// Histogram_Matching_85109071938058
// MI455X (gfx1250) — matrix operand order repaired and checked by reading, not compiled, not run on hardware
//
#include <hip/hip_runtime.h>

typedef __attribute__((ext_vector_type(16))) _Float16 v16h;
typedef __attribute__((ext_vector_type(8)))  float    v8f;

#define NBINS 32
#define FINE  4096
#define IMG_N (1u << 20)   // 1*64*128*128 voxels per image, 2 images

// ---------------------------------------------------------------- kernel 0
__global__ void hm_zero_kernel(unsigned int* __restrict__ p, int n) {
    int i = blockIdx.x * blockDim.x + threadIdx.x;
    if (i < n) p[i] = 0u;
}

// ---------------------------------------------------------------- kernel 1
// Hard histogram with FINE buckets per image (img 0,1 = dst; 2,3 = ref).
// LDS-atomic accumulate, then flush to global with atomics.
__global__ void hm_fine_hist_kernel(const float* __restrict__ dst,
                                    const float* __restrict__ ref,
                                    unsigned int* __restrict__ fineHist) {
    __shared__ unsigned int lh[FINE];
    const int t = threadIdx.x;                       // 256 threads
    for (int i = t; i < FINE; i += 256) lh[i] = 0u;
    __syncthreads();

    const int img = blockIdx.y;                      // 0..3
    const float* src = (img < 2) ? (dst + (size_t)img * IMG_N)
                                 : (ref + (size_t)(img - 2) * IMG_N);
    const float4* s4 = (const float4*)src;
    const int n4 = IMG_N / 4;
    for (int i = blockIdx.x * blockDim.x + t; i < n4; i += gridDim.x * blockDim.x) {
        float4 v = s4[i];
        int j;
        j = min(FINE - 1, max(0, (int)(v.x * (float)FINE))); atomicAdd(&lh[j], 1u);
        j = min(FINE - 1, max(0, (int)(v.y * (float)FINE))); atomicAdd(&lh[j], 1u);
        j = min(FINE - 1, max(0, (int)(v.z * (float)FINE))); atomicAdd(&lh[j], 1u);
        j = min(FINE - 1, max(0, (int)(v.w * (float)FINE))); atomicAdd(&lh[j], 1u);
    }
    __syncthreads();
    for (int i = t; i < FINE; i += 256) {
        unsigned int c = lh[i];
        if (c) atomicAdd(&fineHist[img * FINE + i], c);
    }
}

// ---------------------------------------------------------------- kernel 2
// Soft hist from fine hist -> normalize -> cumsum -> CDF-matching argmin table.
// One block, 512 threads = 16 waves. Wave w: image w>>2, fine-bin quarter w&3.
// Lane = coarse bin (0..31).
__global__ void hm_stats_kernel(const unsigned int* __restrict__ fineHist,
                                float* __restrict__ tables) {
    __shared__ float shist[4][NBINS];
    __shared__ float scdf[4][NBINS];
    const int t = threadIdx.x;
    const int lane = t & 31, wave = t >> 5;
    if (t < 4 * NBINS) shist[t >> 5][t & 31] = 0.0f;
    __syncthreads();

    const int img = wave >> 2, sub = wave & 3;
    const float c = (float)lane;
    float h = 0.0f;
    const int j0 = sub * (FINE / 4);
    for (int j = j0; j < j0 + FINE / 4; ++j) {
        float cnt = (float)fineHist[img * FINE + j];     // uniform per wave
        if (cnt != 0.0f) {
            float q = ((float)j + 0.5f) * (31.0f / (float)FINE);  // bucket center * (BINS-1)
            float d = (q - c) * 5.0f;                              // SIGMA = 5
            float sp = 1.0f / (1.0f + __expf(-(d + 2.5f)));        // DELTA/2*SIGMA = 2.5
            float sm = 1.0f / (1.0f + __expf(-(d - 2.5f)));
            h += cnt * (sp - sm);
        }
    }
    atomicAdd(&shist[img][lane], h);
    __syncthreads();

    if (wave < 4) {
        float hv = shist[wave][lane];
        float tot = hv;
        #pragma unroll
        for (int off = 1; off < 32; off <<= 1) tot += __shfl_xor(tot, off, 32);
        float v = hv / fmaxf(tot, 1e-12f);
        #pragma unroll
        for (int off = 1; off < 32; off <<= 1) {
            float u = __shfl_up(v, off, 32);
            if (lane >= off) v += u;
        }
        scdf[wave][lane] = v;                            // inclusive scan = CDF
    }
    __syncthreads();

    if (wave < 2) {                                      // image pair b = wave
        float myc = scdf[wave][lane];                    // cdf_dst[b][lane]
        int best = 0;
        float bd = fabsf(myc - scdf[2 + wave][0]);
        #pragma unroll
        for (int r = 1; r < NBINS; ++r) {
            float dd = fabsf(myc - scdf[2 + wave][r]);
            if (dd < bd) { bd = dd; best = r; }          // strict < == first argmin
        }
        tables[wave * NBINS + lane] = (float)best;
    }
}

// ---------------------------------------------------------------- kernel 3
// Soft lookup via WMMA, shuffle-free, 2 tiles (32 voxels) per iteration:
//   B (32x16, f16): column n = Gaussian weights of tile-voxel n over k=0..31.
//       Lane n holds K=0..15, lane n+16 holds K=16..31 (contiguous K run ->
//       2 exps + one 16-step geometric recurrence per lane).
//   A (16x32, f16): even rows = lookup table T[k], odd rows = ones.
//   D = A x B: c[0] = numerator, c[1] = denominator of voxel (lane&15),
//   replicated into BOTH lane halves (rows 0/8 and 1/9 identical).
// Pairing two tiles lets lanes 0-15 emit tile0 and lanes 16-31 emit tile1:
// one full-wave coalesced 128B store, no EXEC branch, and two independent
// multiply chains that dual-issue.
__global__ void hm_apply_kernel(const float* __restrict__ dst,
                                const float* __restrict__ tables,
                                float* __restrict__ out) {
    const int lane = threadIdx.x & 31;
    const int L = lane & 15;
    const bool hi = lane >= 16;
    const int wavesPerBlock = blockDim.x >> 5;
    const int gwave = blockIdx.x * wavesPerBlock + (threadIdx.x >> 5);
    const int nwaves = gridDim.x * wavesPerBlock;
    const float R = 0.96078943915f;                     // exp(-2/50)

    for (int b = 0; b < 2; ++b) {
        // ---- A matrix: row L = (L even) ? T[...] : ones (built once per image)
        v16h am;
        if (L & 1) {
            #pragma unroll
            for (int h = 0; h < 16; ++h) am[h] = (_Float16)1.0f;
        } else {
            const float* tb = tables + b * NBINS;
            const int ka = hi ? 16 : 0;                 // halves 0..7 -> K=ka..ka+7
            const int kb = hi ? 24 : 8;                 // halves 8..15 -> K=kb..kb+7
            #pragma unroll
            for (int h = 0; h < 8; ++h) am[h] = (_Float16)tb[ka + h];
            #pragma unroll
            for (int h = 0; h < 8; ++h) am[8 + h] = (_Float16)tb[kb + h];
        }

        const float* img = dst + (size_t)b * IMG_N;
        float* outb = out + (size_t)b * IMG_N;
        const int npairs = IMG_N / 32;
        const float k0 = hi ? 16.0f : 0.0f;

        for (int p = gwave; p < npairs; p += nwaves) {
            const int base = p * 32;
            const float x0 = img[base + L] * 31.0f;        // tile 0, voxel L
            const float x1 = img[base + 16 + L] * 31.0f;   // tile 1, voxel L

            // ---- two B columns: w_k = exp(-(x-k)^2/50), k = k0..k0+15
            float d0 = x0 - k0, d1 = x1 - k0;
            float w0 = __expf(-d0 * d0 * 0.02f);
            float m0 = __expf((2.0f * d0 - 1.0f) * 0.02f);
            float w1 = __expf(-d1 * d1 * 0.02f);
            float m1 = __expf((2.0f * d1 - 1.0f) * 0.02f);
            v16h b0, b1;
            #pragma unroll
            for (int h = 0; h < 16; ++h) {
                b0[h] = (_Float16)w0; w0 *= m0; m0 *= R;
                b1[h] = (_Float16)w1; w1 *= m1; m1 *= R;
            }

            v8f c0 = {}, c1 = {};
            c0 = __builtin_amdgcn_wmma_f32_16x16x32_f16(
                    false, am, false, b0, (short)0, c0, false, false);
            c1 = __builtin_amdgcn_wmma_f32_16x16x32_f16(
                    false, am, false, b1, (short)0, c1, false, false);

            // lanes 0-15 -> tile0 voxel L; lanes 16-31 -> tile1 voxel L
            const float num = hi ? c1[0] : c0[0];
            const float den = hi ? c1[1] : c0[1];
            outb[base + lane] = num * __builtin_amdgcn_rcpf(den) * (1.0f / 31.0f);
        }
    }
}

// ---------------------------------------------------------------- launcher
extern "C" void kernel_launch(void* const* d_in, const int* in_sizes, int n_in,
                              void* d_out, int out_size, void* d_ws, size_t ws_size,
                              hipStream_t stream) {
    const float* dst = (const float*)d_in[0];
    const float* ref = (const float*)d_in[1];
    float* out = (float*)d_out;

    unsigned int* fineHist = (unsigned int*)d_ws;                  // 4 * FINE u32 = 64 KB
    float* tables = (float*)((char*)d_ws + 4 * FINE * sizeof(unsigned int)); // 2*32 f32

    hm_zero_kernel<<<(4 * FINE + 255) / 256, 256, 0, stream>>>(fineHist, 4 * FINE);

    dim3 g1(128, 4);
    hm_fine_hist_kernel<<<g1, 256, 0, stream>>>(dst, ref, fineHist);

    hm_stats_kernel<<<1, 512, 0, stream>>>(fineHist, tables);

    hm_apply_kernel<<<512, 256, 0, stream>>>(dst, tables, out);
}
